// MeanAligning_62311385531121
// MI455X (gfx1250) — compile-verified, matrix operand order checked by reading, not run on hardware
//
#include <hip/hip_runtime.h>
#include <hip/hip_bf16.h>
#include <stdint.h>

// Problem shape (fixed by reference): N*H*W = 16*32*32 positions, K codes, C channels.
#define KDIM 4096
#define CDIM 32
#define NPOS 16384
#define NSEG 4                 // position segments for K2 parallelism
#define SEGP (NPOS / NSEG)     // 4096 positions per segment

typedef __attribute__((ext_vector_type(16))) _Float16     v16h;
typedef __attribute__((ext_vector_type(8)))  float        v8f;
typedef __attribute__((ext_vector_type(4)))  float        f32x4;
typedef __attribute__((ext_vector_type(4)))  int          i32x4v;
typedef __attribute__((ext_vector_type(4)))  unsigned int u32x4;
typedef __attribute__((ext_vector_type(8)))  int          i32x8;
typedef __attribute__((ext_vector_type(4)))  int          i32x4;

#if defined(__has_builtin)
#  if __has_builtin(__builtin_amdgcn_tensor_load_to_lds)
#    define USE_TDM 1
#  endif
#  if __has_builtin(__builtin_amdgcn_ballot_w32)
#    define HAVE_BALLOT32 1
#  endif
#endif

__device__ __forceinline__ int laneId() {
  return __builtin_amdgcn_mbcnt_hi(~0u, __builtin_amdgcn_mbcnt_lo(~0u, 0u));
}

__device__ __forceinline__ unsigned ballot32(bool p) {
#ifdef HAVE_BALLOT32
  return __builtin_amdgcn_ballot_w32(p);
#else
  return (unsigned)__ballot(p);
#endif
}

// wave32 butterfly sum via ds_bpermute (byte index = srcLane*4)
__device__ __forceinline__ float waveSum(float v, int lane) {
#pragma unroll
  for (int m = 16; m >= 1; m >>= 1) {
    int o = __builtin_amdgcn_ds_bpermute(((lane ^ m) << 2),
                                         __builtin_bit_cast(int, v));
    v += __builtin_bit_cast(float, o);
  }
  return v;
}

#ifdef USE_TDM
// Issue one TDM load: 1 x nElem f32 tile at row start `ga` -> LDS offset `lds`.
__device__ __forceinline__ void tdm_load_row(uint64_t ga, uint32_t lds, int nElem) {
  u32x4 g0; i32x8 g1;
  i32x4 g2 = {0, 0, 0, 0}, g3 = {0, 0, 0, 0};
  i32x8 g4 = {0, 0, 0, 0, 0, 0, 0, 0};     // extra group (clang-23 6-arg form)
  g0[0] = 1u;                                            // count=1 (valid user D#)
  g0[1] = lds;                                           // lds_addr (bytes)
  g0[2] = (uint32_t)ga;                                  // global_addr[31:0]
  g0[3] = (uint32_t)((ga >> 32) & 0x1FFFFFFu) | (2u << 30); // addr[56:32] | type=2
  g1[0] = (int)(2u << 16);                               // data_size=4B, no multicast
  g1[1] = (int)((KDIM & 0xFFFFu) << 16);                 // tensor_dim0 lo16
  g1[2] = (int)(((unsigned)NPOS & 0xFFFFu) << 16);       // dim0 hi=0 | tensor_dim1 lo16
  g1[3] = (int)(((unsigned)nElem & 0xFFFFu) << 16);      // dim1 hi=0 | tile_dim0
  g1[4] = 1;                                             // tile_dim1=1, tile_dim2=0
  g1[5] = KDIM;                                          // tensor_dim0_stride lo32
  g1[6] = 0;
  g1[7] = 0;
  __builtin_amdgcn_tensor_load_to_lds(g0, g1, g2, g3, g4, /*cpol=*/0);
}
#endif

// ---------------------------------------------------------------------------
// K1: index extraction. One wave per position: stream code[p, 0..4095] (16 KB)
// and compute idx[p] = sum_k k * code[p,k] (exact for one-hot f32).
// TDM path: two half-row tensor_load_to_lds per wave, split-waited so the DMA
// of half 1 overlaps the FMA scan of half 0.
// ---------------------------------------------------------------------------
__global__ __launch_bounds__(128) void k_extract(const float* __restrict__ code,
                                                 int* __restrict__ idx) {
  const int lane = laneId();
  const int wv   = threadIdx.x >> 5;      // 4 waves / block
  const int p    = blockIdx.x * 4 + wv;   // grid = NPOS/4 blocks
  float s = 0.0f;

#ifdef USE_TDM
  __shared__ float tile[4][KDIM];         // 16 KB per wave, 64 KB per block
  const uint64_t ga   = (uint64_t)(uintptr_t)(code + (size_t)p * KDIM);
  const uint32_t lds0 = (uint32_t)(uintptr_t)(&tile[wv][0]);
  tdm_load_row(ga,                lds0,                KDIM / 2);   // half 0
  tdm_load_row(ga + KDIM * 2,     lds0 + KDIM * 2,     KDIM / 2);   // half 1 (+8 KB)
#  if __has_builtin(__builtin_amdgcn_s_wait_tensorcnt)
  __builtin_amdgcn_s_wait_tensorcnt(1);   // half 0 complete (in-order per wave)
#  else
  asm volatile("s_wait_tensorcnt 0x1" ::: "memory");
#  endif
  const f32x4* t4 = (const f32x4*)(&tile[wv][0]);
#pragma unroll 4
  for (int ch = 0; ch < KDIM / 256; ++ch) {           // elements [0, 2048)
    f32x4 v  = t4[ch * 32 + lane];                    // ds_load_b128
    float c0 = (float)(ch * 128 + lane * 4);
    s += v.x * c0 + v.y * (c0 + 1.0f) + v.z * (c0 + 2.0f) + v.w * (c0 + 3.0f);
  }
#  if __has_builtin(__builtin_amdgcn_s_wait_tensorcnt)
  __builtin_amdgcn_s_wait_tensorcnt(0);   // half 1 complete
#  else
  asm volatile("s_wait_tensorcnt 0x0" ::: "memory");
#  endif
#pragma unroll 4
  for (int ch = KDIM / 256; ch < KDIM / 128; ++ch) {  // elements [2048, 4096)
    f32x4 v  = t4[ch * 32 + lane];
    float c0 = (float)(ch * 128 + lane * 4);
    s += v.x * c0 + v.y * (c0 + 1.0f) + v.z * (c0 + 2.0f) + v.w * (c0 + 3.0f);
  }
#else
  const f32x4* gv = (const f32x4*)(code + (size_t)p * KDIM);
#pragma unroll 4
  for (int ch = 0; ch < KDIM / 128; ++ch) {
    f32x4 v  = gv[ch * 32 + lane];                    // global_load_b128, coalesced
    float c0 = (float)(ch * 128 + lane * 4);
    s += v.x * c0 + v.y * (c0 + 1.0f) + v.z * (c0 + 2.0f) + v.w * (c0 + 3.0f);
  }
#endif
  s = waveSum(s, lane);
  if (lane == 0) idx[p] = (int)(s + 0.5f);
}

// ---------------------------------------------------------------------------
// K2: deterministic per-code partial accumulation (no atomics).
// 256 blocks: block b handles position segment (b&3) for k-rows
// [(b>>2)*64, +64). Indices for the segment are staged chunk-wise into LDS
// with coalesced b128 loads; the 8 waves (8 k-rows each, lane = channel c)
// ballot against low-latency LDS reads in ascending position order.
// Partials are merged in fixed segment order by K3 -> bit-reproducible.
// ---------------------------------------------------------------------------
#define CHUNK 2048
__global__ __launch_bounds__(256) void k_accumulate(const int* __restrict__ idx,
                                                    const float* __restrict__ quant,
                                                    float* __restrict__ num_part,
                                                    int* __restrict__ cnt_part) {
  __shared__ int sidx[CHUNK];             // 8 KB
  const int lane = laneId();
  const int wv   = threadIdx.x >> 5;
  const int seg  = blockIdx.x & 3;
  const int k0   = ((blockIdx.x >> 2) * 8 + wv) * 8;   // 8 k-rows per wave
  const int pbase = seg * SEGP;

  float acc[8];
  int   cnt[8];
#pragma unroll
  for (int r = 0; r < 8; ++r) { acc[r] = 0.0f; cnt[r] = 0; }

  for (int cb = 0; cb < SEGP; cb += CHUNK) {
    // stage CHUNK indices into LDS: 256 threads x 2 int4, fully coalesced
    {
      const i32x4v* g4 = (const i32x4v*)(idx + pbase + cb);
      i32x4v* s4 = (i32x4v*)sidx;
      s4[threadIdx.x]       = g4[threadIdx.x];
      s4[threadIdx.x + 256] = g4[threadIdx.x + 256];
    }
    __syncthreads();
    for (int sub = 0; sub < CHUNK; sub += 32) {
      int myidx = sidx[sub + lane];
#pragma unroll
      for (int r = 0; r < 8; ++r) {
        unsigned mb = ballot32(myidx == (k0 + r));
        cnt[r] += __popc(mb);
        while (mb) {                       // ascending position order
          int b = __ffs(mb) - 1;
          mb &= mb - 1;
          acc[r] += quant[(size_t)(pbase + cb + sub + b) * CDIM + lane];
        }
      }
    }
    __syncthreads();
  }
#pragma unroll
  for (int r = 0; r < 8; ++r) {
    num_part[((size_t)seg * KDIM + (k0 + r)) * CDIM + lane] = acc[r];
    if (lane == 0) cnt_part[seg * KDIM + k0 + r] = cnt[r];
  }
}

// ---------------------------------------------------------------------------
// K3: masked sum-of-squares via WMMA. Per 16-row k-tile: A = sq diffs in the
// 16-bit A-matrix layout (lane -> row M=lane%16; 16 halfs cover that lane's
// K-columns), B = all-ones (layout invariant) => D[m,*] = sum_c sq[m,c].
// sq split into f16 hi+lo terms, two chained WMMAs -> ~2^-22 rel accuracy.
// Segment partials merged in fixed order (deterministic).
// ---------------------------------------------------------------------------
__global__ __launch_bounds__(256) void k_loss(const float* __restrict__ codebook,
                                              const float* __restrict__ num_part,
                                              const int* __restrict__ cnt_part,
                                              float* __restrict__ partial) {
  __shared__ float redL[8], redV[8];
  const int lane  = laneId();
  const int wv    = threadIdx.x >> 5;
  const int m     = lane & 15;
  const int base0 = (lane < 16) ? 0 : 8;

  v8f acc = {0.f, 0.f, 0.f, 0.f, 0.f, 0.f, 0.f, 0.f};
  v16h ones;
#pragma unroll
  for (int j = 0; j < 16; ++j) ones[j] = (_Float16)1.0f;
  float nval = 0.0f;

  // 256 k-tiles total; 8 blocks * 8 waves, 4 tiles per wave.
  for (int t = 0; t < 4; ++t) {
    const int tile = (blockIdx.x * 8 + wv) * 4 + t;
    const int k    = tile * 16 + m;
    int c_ = 0;
#pragma unroll
    for (int sg = 0; sg < NSEG; ++sg) c_ += cnt_part[sg * KDIM + k];
    const float inv = (c_ > 0) ? (1.0f / (float)c_) : 0.0f;
    v16h hi, lo;
#pragma unroll
    for (int j = 0; j < 16; ++j) {
      const int c = base0 + j + ((j >= 8) ? 8 : 0);    // ISA 16-bit A layout
      float nsum = 0.0f;
#pragma unroll
      for (int sg = 0; sg < NSEG; ++sg)                // fixed merge order
        nsum += num_part[((size_t)sg * KDIM + k) * CDIM + c];
      const float mean = nsum * inv;
      const float d    = codebook[(size_t)k * CDIM + c] - mean;
      const float sq   = (c_ > 0) ? d * d : 0.0f;
      const _Float16 h = (_Float16)sq;
      hi[j] = h;
      lo[j] = (_Float16)(sq - (float)h);
    }
    acc = __builtin_amdgcn_wmma_f32_16x16x32_f16(false, lo, false, ones,
                                                 (short)0, acc, false, false);
    acc = __builtin_amdgcn_wmma_f32_16x16x32_f16(false, hi, false, ones,
                                                 (short)0, acc, false, false);
    if (lane < 16 && c_ > 0) nval += 1.0f;
  }

  // D layout: VGPR r, lane 0 -> row r; lane 16 -> row r+8 (column N=0).
  float s = 0.0f;
  if (lane == 0 || lane == 16) {
#pragma unroll
    for (int r = 0; r < 8; ++r) s += acc[r];
  }
  s    = waveSum(s, lane);
  nval = waveSum(nval, lane);
  if (lane == 0) { redL[wv] = s; redV[wv] = nval; }
  __syncthreads();
  if (threadIdx.x == 0) {
    float L = 0.0f, V = 0.0f;
    for (int i = 0; i < 8; ++i) { L += redL[i]; V += redV[i]; }  // fixed order
    partial[blockIdx.x * 2 + 0] = L;
    partial[blockIdx.x * 2 + 1] = V;
  }
}

__global__ void k_final(const float* __restrict__ partial, float* __restrict__ out) {
  if (threadIdx.x == 0 && blockIdx.x == 0) {
    float L = 0.0f, V = 0.0f;
    for (int i = 0; i < 8; ++i) { L += partial[2 * i]; V += partial[2 * i + 1]; }
    if (V < 1.0f) V = 1.0f;
    out[0] = L / (V * (float)CDIM);
  }
}

extern "C" void kernel_launch(void* const* d_in, const int* in_sizes, int n_in,
                              void* d_out, int out_size, void* d_ws, size_t ws_size,
                              hipStream_t stream) {
  (void)in_sizes; (void)n_in; (void)out_size; (void)ws_size;
  const float* quant    = (const float*)d_in[0];  // [16,32,32,32] f32
  const float* code     = (const float*)d_in[1];  // [16,32,32,4096] f32 one-hot
  const float* codebook = (const float*)d_in[2];  // [4096,32] f32

  char*  ws       = (char*)d_ws;
  int*   idx      = (int*)(ws);                        //  64 KB
  int*   cnt_part = (int*)(ws + 65536);                //  64 KB (4 segs x 4096)
  float* num_part = (float*)(ws + 131072);             //   2 MB (4 segs x 4096 x 32)
  float* partial  = (float*)(ws + 131072 + 2097152);   //  64 B

  k_extract   <<<NPOS / 4, 128, 0, stream>>>(code, idx);
  k_accumulate<<<256,      256, 0, stream>>>(idx, quant, num_part, cnt_part);
  k_loss      <<<8,        256, 0, stream>>>(codebook, num_part, cnt_part, partial);
  k_final     <<<1,         32, 0, stream>>>(partial, (float*)d_out);
}
